// Cwt_65816078843929
// MI455X (gfx1250) — matrix operand order repaired and checked by reading, not compiled, not run on hardware
//
#include <hip/hip_runtime.h>

typedef __attribute__((ext_vector_type(16))) _Float16 v16h;
typedef __attribute__((ext_vector_type(8)))  _Float16 v8h;
typedef __attribute__((ext_vector_type(8)))  float    v8f;

#define B_        128
#define N_        1024
#define S_        101
#define SP_       112      // scales padded to 7 WMMA tiles
#define XS_STRIDE 2048
#define OUT_H     200
#define OUT_W     200

// ---- workspace layout (bytes) ----
// krev : f16 [SP_][N_]                 at 0        (229,376 B)
// xs   : f16 [8][B_][XS_STRIDE]        at 229376   (4,194,304 B)
// res  : f32 [B_][S_][N_]              at 4423680  (52,953,088 B)
// mnmx : f32 [B_][2]                   at 57376768 (1,024 B)

// Krev[s][v] = kernels[s][N-1-v]  (reversed kernel), zero-padded for s >= 101
__global__ void prep_krev(const float* __restrict__ kern, _Float16* __restrict__ krev) {
    int idx = blockIdx.x * blockDim.x + threadIdx.x;
    if (idx >= SP_ * N_) return;
    int s = idx >> 10, v = idx & (N_ - 1);
    float val = (s < S_) ? kern[s * N_ + (N_ - 1 - v)] : 0.0f;
    krev[idx] = (_Float16)val;
}

// xs[sh][b][i] = x_pad[b][i+sh], x_pad[b][j] = x[b][j-511] for j in [511,1535), else 0
__global__ void prep_xs(const float* __restrict__ x, _Float16* __restrict__ xs) {
    int idx = blockIdx.x * blockDim.x + threadIdx.x;
    if (idx >= 8 * B_ * XS_STRIDE) return;
    int i  = idx & (XS_STRIDE - 1);
    int b  = (idx >> 11) & (B_ - 1);
    int sh = idx >> 18;
    int j  = i + sh;
    float val = (j >= 511 && j < 511 + N_) ? x[b * N_ + (j - 511)] : 0.0f;
    xs[idx] = (_Float16)val;
}

// GEMM: res[b][s][w] = sum_v Krev[s][v] * x_pad[b][w+v]
// block: 224 threads = 7 waves; wave = one s-tile (16 scales) x 128 w (8 w-tiles)
// grid.x = B_ * 8 (b, w-block of 128)
__global__ __launch_bounds__(224) void cwt_wmma(const _Float16* __restrict__ xs,
                                                const _Float16* __restrict__ krev,
                                                float* __restrict__ res) {
    const int b    = blockIdx.x >> 3;
    const int wb   = blockIdx.x & 7;
    const int st   = threadIdx.x >> 5;     // s-tile 0..6
    const int lane = threadIdx.x & 31;
    const int n    = lane & 15;
    const int half = lane >> 4;
    const int koffB = half << 3;           // B-frag K offset: 0 / 8
    const int kbA   = half << 3;           // A-frag K base:   0 / 8
    const int shift = n & 7;
    const int nb8   = n & 8;

    const _Float16* __restrict__ xrow = xs + (size_t)(shift * B_ + b) * XS_STRIDE;
    const _Float16* __restrict__ arow = krev + (size_t)(st * 16 + n) * N_;

    v8f acc[8];
    v8f zero = {0.f,0.f,0.f,0.f,0.f,0.f,0.f,0.f};
#pragma unroll
    for (int i = 0; i < 8; ++i) acc[i] = zero;

    const int wbase0 = wb * 128 + nb8 + koffB;   // all terms ≡ 0 mod 8 -> 16B aligned

    for (int kk = 0; kk < N_; kk += 32) {
        // A fragment: Krev row (st*16 + n), K chunks [kk+kbA..+7], [kk+16+kbA..+7]
        v8h a0 = *(const v8h*)(arow + kk + kbA);
        v8h a1 = *(const v8h*)(arow + kk + 16 + kbA);
        v16h afrag = __builtin_shufflevector(a0, a1,
            0,1,2,3,4,5,6,7,8,9,10,11,12,13,14,15);
#pragma unroll
        for (int wt = 0; wt < 8; ++wt) {
            const _Float16* bp = xrow + (wbase0 + wt * 16 + kk);
            v8h b0 = *(const v8h*)(bp);
            v8h b1 = *(const v8h*)(bp + 16);
            v16h bfrag = __builtin_shufflevector(b0, b1,
                0,1,2,3,4,5,6,7,8,9,10,11,12,13,14,15);
            acc[wt] = __builtin_amdgcn_wmma_f32_16x16x32_f16(
                false, afrag, false, bfrag, (short)0, acc[wt], false, false);
        }
    }

    // D layout: vgpr r, lane l -> M = r + 8*half (scale), N = l&15 (w)
#pragma unroll
    for (int wt = 0; wt < 8; ++wt) {
        int w = wb * 128 + wt * 16 + n;
#pragma unroll
        for (int r = 0; r < 8; ++r) {
            int s = st * 16 + r + (half << 3);
            if (s < S_) {
                res[((size_t)b * S_ + s) * N_ + w] = acc[wt][r];
            }
        }
    }
}

// per-batch min/max over res[b][0..100][0..1023]
__global__ void minmax_kernel(const float* __restrict__ res, float* __restrict__ mnmx) {
    __shared__ float smn[256], smx[256];
    int b = blockIdx.x;
    const float* p = res + (size_t)b * S_ * N_;
    float mn = 3.4e38f, mx = -3.4e38f;
    for (int i = threadIdx.x; i < S_ * N_; i += 256) {
        float v = p[i];
        mn = fminf(mn, v);
        mx = fmaxf(mx, v);
    }
    smn[threadIdx.x] = mn;
    smx[threadIdx.x] = mx;
    __syncthreads();
    for (int off = 128; off > 0; off >>= 1) {
        if (threadIdx.x < off) {
            smn[threadIdx.x] = fminf(smn[threadIdx.x], smn[threadIdx.x + off]);
            smx[threadIdx.x] = fmaxf(smx[threadIdx.x], smx[threadIdx.x + off]);
        }
        __syncthreads();
    }
    if (threadIdx.x == 0) {
        mnmx[b * 2 + 0] = smn[0];
        mnmx[b * 2 + 1] = smx[0];
    }
}

// fused normalize + bilinear resize (101,1024) -> (200,200), half-pixel centers
__global__ void resize_kernel(const float* __restrict__ res, const float* __restrict__ mnmx,
                              float* __restrict__ out) {
    int idx = blockIdx.x * blockDim.x + threadIdx.x;
    if (idx >= B_ * OUT_H * OUT_W) return;
    int j = idx % OUT_W;
    int i = (idx / OUT_W) % OUT_H;
    int b = idx / (OUT_W * OUT_H);
    float sh = ((float)i + 0.5f) * ((float)S_ / (float)OUT_H) - 0.5f;
    float sw = ((float)j + 0.5f) * ((float)N_ / (float)OUT_W) - 0.5f;
    sh = fminf(fmaxf(sh, 0.0f), (float)(S_ - 1));
    sw = fminf(fmaxf(sw, 0.0f), (float)(N_ - 1));
    int h0 = (int)sh, w0 = (int)sw;
    int h1 = min(h0 + 1, S_ - 1), w1 = min(w0 + 1, N_ - 1);
    float th = sh - (float)h0, tw = sw - (float)w0;
    const float* p = res + (size_t)b * S_ * N_;
    float v00 = p[h0 * N_ + w0], v01 = p[h0 * N_ + w1];
    float v10 = p[h1 * N_ + w0], v11 = p[h1 * N_ + w1];
    float v0 = v00 + (v01 - v00) * tw;
    float v1 = v10 + (v11 - v10) * tw;
    float v  = v0 + (v1 - v0) * th;
    float mn = mnmx[b * 2 + 0], mx = mnmx[b * 2 + 1];
    out[idx] = (v - mn) / (mx - mn);
}

extern "C" void kernel_launch(void* const* d_in, const int* in_sizes, int n_in,
                              void* d_out, int out_size, void* d_ws, size_t ws_size,
                              hipStream_t stream) {
    const float* x    = (const float*)d_in[0];   // (128, 1024) f32
    const float* kern = (const float*)d_in[1];   // (101, 1024) f32
    char* ws = (char*)d_ws;

    _Float16* krev = (_Float16*)(ws);
    _Float16* xs   = (_Float16*)(ws + 229376);
    float*    res  = (float*)(ws + 4423680);
    float*    mnmx = (float*)(ws + 57376768);
    float*    out  = (float*)d_out;

    prep_krev<<<(SP_ * N_ + 255) / 256, 256, 0, stream>>>(kern, krev);
    prep_xs<<<(8 * B_ * XS_STRIDE + 255) / 256, 256, 0, stream>>>(x, xs);
    cwt_wmma<<<B_ * 8, 224, 0, stream>>>(xs, krev, res);
    minmax_kernel<<<B_, 256, 0, stream>>>(res, mnmx);
    resize_kernel<<<(B_ * OUT_H * OUT_W + 255) / 256, 256, 0, stream>>>(res, mnmx, out);
}
